// mhsa_62560493633795
// MI455X (gfx1250) — matrix operand order repaired and checked by reading, not compiled, not run on hardware
//
#include <hip/hip_runtime.h>

#define B_  2
#define S_  2048
#define D_  1024
#define H_  16
#define DH_ 64
#define TN  64          // keys per LDS tile
#define VT_STRIDE 72    // padded stride for transposed V in LDS

typedef __attribute__((ext_vector_type(16))) __bf16 v16bf;
typedef __attribute__((ext_vector_type(8)))  __bf16 v8bf;
typedef __attribute__((ext_vector_type(8)))  float  v8f;
typedef __attribute__((ext_vector_type(4)))  int    i4;

#if defined(__has_builtin)
#if __has_builtin(__builtin_amdgcn_global_load_async_to_lds_b128)
#define HAVE_ASYNC_LDS 1
#endif
#endif

__device__ __forceinline__ v16bf cat16(v8bf a, v8bf b) {
  return __builtin_shufflevector(a, b, 0,1,2,3,4,5,6,7,8,9,10,11,12,13,14,15);
}

// packed f32->bf16 conversion (lowers to v_cvt_pk_bf16_f32 class ops)
__device__ __forceinline__ v8bf cvt8(v8f f) {
  return __builtin_convertvector(f, v8bf);
}

__device__ __forceinline__ v8f wmma_bf16(v16bf a, v16bf b, v8f c) {
  return __builtin_amdgcn_wmma_f32_16x16x32_bf16(false, a, false, b, (short)0, c,
                                                 false, false);
}

// ---------------------------------------------------------------------------
// Kernel 1: per-head QKV projection. One wave computes a 16-row tile of
// Q,K,V for one (b,h). Q is pre-scaled by 1/sqrt(DH). Output bf16.
// ---------------------------------------------------------------------------
__global__ __launch_bounds__(256) void qkv_proj_kernel(
    const float* __restrict__ x,
    const float* __restrict__ Wq, const float* __restrict__ bq,
    const float* __restrict__ Wk, const float* __restrict__ bk,
    const float* __restrict__ Wv, const float* __restrict__ bv,
    __bf16* __restrict__ Qg, __bf16* __restrict__ Kg, __bf16* __restrict__ Vg)
{
  const int lane = threadIdx.x & 31;
  const int half = lane >> 4;
  const int mrow = lane & 15;
  const int w    = blockIdx.x * 8 + (threadIdx.x >> 5);   // global wave id
  const int bh   = w >> 7;                                // S/16 = 128 tiles per (b,h)
  const int b    = bh >> 4;
  const int h    = bh & 15;
  const int s0   = (w & 127) << 4;

  // A fragments (16x32 bf16) of X for this head: K-dim = feature d (0..63)
  const float* xrow = x + ((size_t)b * S_ + s0 + mrow) * D_ + h * DH_;
  v16bf a[2];
  #pragma unroll
  for (int kk = 0; kk < 2; ++kk) {
    const int d0 = kk * 32 + half * 8;
    v8f x0 = *(const v8f*)(xrow + d0);        // elements 0..7  : K=d0..d0+7
    v8f x1 = *(const v8f*)(xrow + d0 + 16);   // elements 8..15 : K=d0+16..d0+23
    a[kk] = cat16(cvt8(x0), cvt8(x1));
  }

  const size_t wofs = (size_t)h * DH_ * DH_;
  const float* Wm[3] = { Wq + wofs, Wk + wofs, Wv + wofs };
  const float* bm[3] = { bq + h * DH_, bk + h * DH_, bv + h * DH_ };
  const size_t oofs = ((size_t)(b * H_ + h) * S_ + s0) * DH_;
  __bf16* om[3] = { Qg + oofs, Kg + oofs, Vg + oofs };
  const float sm[3] = { 0.125f, 1.0f, 1.0f };   // 1/sqrt(64) folded into Q

  #pragma unroll
  for (int m = 0; m < 3; ++m) {
    #pragma unroll
    for (int nt = 0; nt < 4; ++nt) {
      const int e = nt * 16 + mrow;               // output feature (B-frag N)
      v8f acc = {};
      #pragma unroll
      for (int kk = 0; kk < 2; ++kk) {
        // B-frag: input features d0..d0+7 and d0+16..d0+23 of W[h][e][*]
        const float* wr = Wm[m] + (size_t)e * DH_ + kk * 32 + half * 8;
        v8f w0 = *(const v8f*)(wr);
        v8f w1 = *(const v8f*)(wr + 16);
        acc = wmma_bf16(a[kk], cat16(cvt8(w0), cvt8(w1)), acc);
      }
      const float bias = bm[m][e];
      v8f t;
      #pragma unroll
      for (int j = 0; j < 8; ++j) t[j] = (acc[j] + bias) * sm[m];
      v8bf ob = cvt8(t);
      #pragma unroll
      for (int j = 0; j < 8; ++j)
        om[m][(size_t)(j + 8 * half) * DH_ + e] = ob[j];   // row M=j+8*half
    }
  }
}

// ---------------------------------------------------------------------------
// Kernel 2: flash attention. Block = 256 threads = 8 waves; each wave owns a
// 16-row query tile (block covers 128 rows). Streams 64-key K/V tiles via
// LDS; online softmax in fp32; P staged through per-wave LDS to convert the
// C-layout scores into A-layout bf16 fragments for the P@V WMMAs.
// ---------------------------------------------------------------------------
__global__ __launch_bounds__(256) void flash_attn_kernel(
    const __bf16* __restrict__ Qg, const __bf16* __restrict__ Kg,
    const __bf16* __restrict__ Vg, float* __restrict__ out)
{
  __shared__ __attribute__((aligned(16))) __bf16 Ks[TN * DH_];        // [key][feat]
  __shared__ __attribute__((aligned(16))) __bf16 Vt[DH_ * VT_STRIDE]; // [feat][key]
  __shared__ __attribute__((aligned(16))) __bf16 Ps[8 * 16 * TN];     // per-wave P

  const int tid  = threadIdx.x;
  const int wave = tid >> 5;
  const int lane = tid & 31;
  const int half = lane >> 4;
  const int mrow = lane & 15;
  const int b = blockIdx.z, h = blockIdx.y;
  const int s0 = blockIdx.x * 128 + wave * 16;

  const size_t bhOff = (size_t)(b * H_ + h) * S_ * DH_;
  const __bf16* Qbh = Qg + bhOff;
  const __bf16* Kbh = Kg + bhOff;
  const __bf16* Vbh = Vg + bhOff;

  // Q A-fragments stay in registers for the whole pass (scale pre-applied)
  v16bf qa[2];
  {
    const __bf16* qrow = Qbh + (size_t)(s0 + mrow) * DH_;
    #pragma unroll
    for (int kk = 0; kk < 2; ++kk) {
      const int d0 = kk * 32 + half * 8;
      qa[kk] = cat16(*(const v8bf*)(qrow + d0),
                     *(const v8bf*)(qrow + d0 + 16));
    }
  }

  v8f o[4];
  #pragma unroll
  for (int fn = 0; fn < 4; ++fn) o[fn] = {};
  float mrun[8], lrun[8];
  #pragma unroll
  for (int j = 0; j < 8; ++j) { mrun[j] = -1e30f; lrun[j] = 0.f; }

  const int r = tid >> 2;          // 0..63 : tile row this thread loads
  const int c = (tid & 3) << 4;    // 0,16,32,48 : feature chunk
  __bf16* pw = Ps + wave * (16 * TN);

  for (int t = 0; t < S_; t += TN) {
    // ---- cooperative K / V(transposed) tile load -------------------------
    {
      const __bf16* krow = Kbh + (size_t)(t + r) * DH_ + c;
#if HAVE_ASYNC_LDS
      // CDNA5 async copy: per-lane 16B global -> LDS, tracked by ASYNCcnt
      __builtin_amdgcn_global_load_async_to_lds_b128(
          (__attribute__((address_space(1))) i4*)(krow),
          (__attribute__((address_space(3))) i4*)(Ks + r * DH_ + c), 0, 0);
      __builtin_amdgcn_global_load_async_to_lds_b128(
          (__attribute__((address_space(1))) i4*)(krow + 8),
          (__attribute__((address_space(3))) i4*)(Ks + r * DH_ + c + 8), 0, 0);
#else
      *(v8bf*)(Ks + r * DH_ + c)     = *(const v8bf*)(krow);
      *(v8bf*)(Ks + r * DH_ + c + 8) = *(const v8bf*)(krow + 8);
#endif
      const __bf16* vrow = Vbh + (size_t)(t + r) * DH_ + c;
      v8bf vv0 = *(const v8bf*)(vrow);
      v8bf vv1 = *(const v8bf*)(vrow + 8);
      #pragma unroll
      for (int i = 0; i < 8; ++i) Vt[(c + i) * VT_STRIDE + r]     = vv0[i];
      #pragma unroll
      for (int i = 0; i < 8; ++i) Vt[(c + 8 + i) * VT_STRIDE + r] = vv1[i];
    }
#if HAVE_ASYNC_LDS
    asm volatile("s_wait_asynccnt 0" ::: "memory");  // drain own async copies
#endif
    __syncthreads();

    // ---- scores: S = Q @ K^T  (4 key subtiles x 2 K-chunks) --------------
    v8f sc[4];
    #pragma unroll
    for (int nt = 0; nt < 4; ++nt) {
      v8f accs = {};
      #pragma unroll
      for (int kk = 0; kk < 2; ++kk) {
        const __bf16* kp = Ks + (nt * 16 + mrow) * DH_ + kk * 32 + half * 8;
        accs = wmma_bf16(qa[kk],
                         cat16(*(const v8bf*)kp, *(const v8bf*)(kp + 16)),
                         accs);
      }
      sc[nt] = accs;
    }

    // ---- online softmax (row reductions over 16-lane half-groups) --------
    #pragma unroll
    for (int j = 0; j < 8; ++j) {
      float mx = fmaxf(fmaxf(sc[0][j], sc[1][j]), fmaxf(sc[2][j], sc[3][j]));
      #pragma unroll
      for (int off = 1; off < 16; off <<= 1)
        mx = fmaxf(mx, __shfl_xor(mx, off, 32));
      const float mnew  = fmaxf(mrun[j], mx);
      const float alpha = __expf(mrun[j] - mnew);
      mrun[j] = mnew;
      float rsum = 0.f;
      #pragma unroll
      for (int nt = 0; nt < 4; ++nt) {
        const float p = __expf(sc[nt][j] - mnew);
        sc[nt][j] = p;
        rsum += p;
      }
      #pragma unroll
      for (int off = 1; off < 16; off <<= 1)
        rsum += __shfl_xor(rsum, off, 32);
      lrun[j] = lrun[j] * alpha + rsum;
      #pragma unroll
      for (int fn = 0; fn < 4; ++fn) o[fn][j] *= alpha;
    }

    // ---- stage P (C-layout -> row-major 16xTN in per-wave LDS) -----------
    #pragma unroll
    for (int nt = 0; nt < 4; ++nt) {
      v8bf pb = cvt8(sc[nt]);                    // one packed convert per tile
      #pragma unroll
      for (int j = 0; j < 8; ++j)
        pw[(j + 8 * half) * TN + nt * 16 + mrow] = pb[j];
    }

    // wave-local LDS ordering: same-wave DS ops are in-order; just drain DScnt
    asm volatile("s_wait_dscnt 0" ::: "memory");

    // ---- reload P as A-fragments -----------------------------------------
    v16bf pa[2];
    {
      const __bf16* prow = pw + mrow * TN;
      #pragma unroll
      for (int kk = 0; kk < 2; ++kk) {
        const int d0 = kk * 32 + half * 8;
        pa[kk] = cat16(*(const v8bf*)(prow + d0),
                       *(const v8bf*)(prow + d0 + 16));
      }
    }

    // ---- O += P @ V (B-frags contiguous thanks to transposed V) ----------
    #pragma unroll
    for (int fn = 0; fn < 4; ++fn) {
      v8f acc = o[fn];
      #pragma unroll
      for (int kk = 0; kk < 2; ++kk) {
        const __bf16* vp = Vt + (fn * 16 + mrow) * VT_STRIDE + kk * 32 + half * 8;
        acc = wmma_bf16(pa[kk],
                        cat16(*(const v8bf*)vp, *(const v8bf*)(vp + 16)),
                        acc);
      }
      o[fn] = acc;
    }
    __syncthreads();   // protect Ks/Vt before next tile overwrite
  }

  // ---- normalize and store fp32 output [B,S,D] ---------------------------
  #pragma unroll
  for (int fn = 0; fn < 4; ++fn) {
    #pragma unroll
    for (int j = 0; j < 8; ++j) {
      const int M = j + 8 * half;
      out[((size_t)b * S_ + s0 + M) * D_ + h * DH_ + fn * 16 + mrow] =
          o[fn][j] / lrun[j];
    }
  }
}

// ---------------------------------------------------------------------------
extern "C" void kernel_launch(void* const* d_in, const int* in_sizes, int n_in,
                              void* d_out, int out_size, void* d_ws, size_t ws_size,
                              hipStream_t stream) {
  const float* x  = (const float*)d_in[0];
  const float* Wq = (const float*)d_in[1];
  const float* bq = (const float*)d_in[2];
  const float* Wk = (const float*)d_in[3];
  const float* bk = (const float*)d_in[4];
  const float* Wv = (const float*)d_in[5];
  const float* bv = (const float*)d_in[6];
  float* out = (float*)d_out;

  const size_t N = (size_t)B_ * H_ * S_ * DH_;   // elements per Q/K/V buffer
  __bf16* Qg = (__bf16*)d_ws;                    // 3*N*2 bytes ≈ 25 MB of d_ws
  __bf16* Kg = Qg + N;
  __bf16* Vg = Kg + N;

  // 4096 waves total / 8 waves per block
  qkv_proj_kernel<<<dim3((B_ * H_ * (S_ / 16)) / 8), 256, 0, stream>>>(
      x, Wq, bq, Wk, bk, Wv, bv, Qg, Kg, Vg);

  flash_attn_kernel<<<dim3(S_ / 128, H_, B_), 256, 0, stream>>>(Qg, Kg, Vg, out);
}
